// ChunkedPCLayer_47828755808851
// MI455X (gfx1250) — matrix operand order repaired and checked by reading, not compiled, not run on hardware
//
#include <hip/hip_runtime.h>
#include <math.h>

typedef __bf16 bf16;
typedef __attribute__((ext_vector_type(16))) __bf16 v16bf;
typedef __attribute__((ext_vector_type(8)))  float  v8f;
typedef int v4i __attribute__((ext_vector_type(4)));

#define TOKENS 16384   // B*N = 4*4096
#define EMB    1024
#define FFD    4096
#define HEADS  16
#define HD     64
#define CH     64      // chunk size
#define WIN    128     // window = 2*CH
#define SEQ    4096

// ---------------------------------------------------------------------------
// gfx1250 async global->LDS copy (ASYNCcnt path), with portable fallback
// ---------------------------------------------------------------------------
__device__ inline void async_copy16(const bf16* g, bf16* l) {
#if defined(__gfx1250__) && __has_builtin(__builtin_amdgcn_global_load_async_to_lds_b128)
  __builtin_amdgcn_global_load_async_to_lds_b128(
      (__attribute__((address_space(1))) v4i*)g,
      (__attribute__((address_space(3))) v4i*)l, 0, 0);
#else
  *(uint4*)l = *(const uint4*)g;
#endif
}

__device__ inline void wait_async() {
#if defined(__gfx1250__) && __has_builtin(__builtin_amdgcn_s_wait_asynccnt)
  __builtin_amdgcn_s_wait_asynccnt(0);
#endif
}

// ---------------------------------------------------------------------------
// WMMA fragment loaders (wave32; contiguous K order, same in A and B)
// A 16x32 bf16: lane L (row = L&15): hi=L>>4 -> K = hi*16..hi*16+15,
//                                              contiguous K per lane.
// B 32x16 bf16: lane L (col = L&15): lanes<16 K=0..15, lanes>=16 K=16..31,
//               contiguous K per lane.
// ---------------------------------------------------------------------------
__device__ inline v16bf load_frag_a(const bf16* base, int stride, int row0,
                                    int k0, int lane) {
  int r = lane & 15, hi = lane >> 4;
  union { v16bf v; uint4 q[2]; } u;
  const bf16* p = base + (size_t)(row0 + r) * stride + k0;
  u.q[0] = *(const uint4*)(p + hi * 16);
  u.q[1] = *(const uint4*)(p + hi * 16 + 8);
  return u.v;
}

__device__ inline v16bf load_frag_b(const bf16* base, int stride, int col0,
                                    int k0, int lane) {
  int c = lane & 15, hi = lane >> 4;
  union { v16bf v; uint4 q[2]; } u;
  const bf16* p = base + (size_t)(col0 + c) * stride + k0 + hi * 16;
  u.q[0] = *(const uint4*)(p);
  u.q[1] = *(const uint4*)(p + 8);
  return u.v;
}

// ---------------------------------------------------------------------------
// f32 -> bf16 conversion
// ---------------------------------------------------------------------------
__global__ __launch_bounds__(256) void convert_bf16_kernel(
    const float* __restrict__ src, bf16* __restrict__ dst, size_t n) {
  size_t i = (size_t)blockIdx.x * blockDim.x + threadIdx.x;
  size_t stride = (size_t)gridDim.x * blockDim.x;
  for (; i < n; i += stride) dst[i] = (bf16)src[i];
}

// ---------------------------------------------------------------------------
// LayerNorm over E=1024, one block (256 thr) per row, bf16 output
// ---------------------------------------------------------------------------
__global__ __launch_bounds__(256) void ln_kernel(
    const float* __restrict__ x, const float* __restrict__ g,
    const float* __restrict__ b, bf16* __restrict__ out) {
  __shared__ float red[256];
  int row = blockIdx.x, t = threadIdx.x;
  const float* xr = x + (size_t)row * EMB;
  float v[4], s = 0.f;
#pragma unroll
  for (int i = 0; i < 4; ++i) { v[i] = xr[t + 256 * i]; s += v[i]; }
  red[t] = s; __syncthreads();
  for (int o = 128; o > 0; o >>= 1) {
    if (t < o) red[t] += red[t + o];
    __syncthreads();
  }
  float mean = red[0] * (1.f / EMB);
  __syncthreads();
  float vs = 0.f;
#pragma unroll
  for (int i = 0; i < 4; ++i) { float d = v[i] - mean; vs += d * d; }
  red[t] = vs; __syncthreads();
  for (int o = 128; o > 0; o >>= 1) {
    if (t < o) red[t] += red[t + o];
    __syncthreads();
  }
  float rstd = rsqrtf(red[0] * (1.f / EMB) + 1e-5f);
  bf16* orow = out + (size_t)row * EMB;
#pragma unroll
  for (int i = 0; i < 4; ++i) {
    int c = t + 256 * i;
    orow[c] = (bf16)((v[i] - mean) * rstd * g[c] + b[c]);
  }
}

// ---------------------------------------------------------------------------
// Tiled WMMA GEMM: out[M,N] = A[M,K](bf16) @ W[K,N](bf16) + bias (+epilogue)
// 128x128 tile, BK=32, double-buffered LDS; A tile via async global->LDS.
// 8 waves: wave = (wm 0..3) x (wn 0..1) -> 32x64 region, 8 WMMA / K-step.
// ---------------------------------------------------------------------------
enum { EPI_BF16_BIAS = 0, EPI_F32_BIAS_RES = 1, EPI_BF16_BIAS_GELU = 2 };

template <int EPI>
__global__ __launch_bounds__(256) void gemm_bf16_kernel(
    const bf16* __restrict__ A, const bf16* __restrict__ W,
    const float* __restrict__ bias, const float* __restrict__ resid,
    void* __restrict__ outv, int M, int N, int K) {
  __shared__ bf16 As[2][128 * 32];
  __shared__ bf16 Bs[2][128 * 32];
  int t = threadIdx.x, lane = t & 31, w = t >> 5;
  int m0 = blockIdx.y * 128, n0 = blockIdx.x * 128;
  int wm = w >> 1, wn = w & 1;

  v8f acc[2][4];
#pragma unroll
  for (int mi = 0; mi < 2; ++mi)
#pragma unroll
    for (int ni = 0; ni < 4; ++ni) acc[mi][ni] = v8f{};

  int ar = t >> 1, ah = (t & 1) * 16;       // A: 2 thr/row, 16 bf16 each
  int bk = t >> 3, bn = (t & 7) * 16;       // B: 1 k-row, 16 n-cols each

  const bf16* arow = A + (size_t)(m0 + ar) * K + ah;

  auto stage_tile = [&](int k0, int buf) {
    // A tile: contiguous 32B/lane -> async global->LDS
    async_copy16(arow + k0, &As[buf][ar * 32 + ah]);
    async_copy16(arow + k0 + 8, &As[buf][ar * 32 + ah + 8]);
    // B tile: transpose through VGPRs into [n][k] layout
    union { uint4 q[2]; bf16 e[16]; } u;
    const uint4* bsrc = (const uint4*)(W + (size_t)(k0 + bk) * N + n0 + bn);
    u.q[0] = bsrc[0]; u.q[1] = bsrc[1];
#pragma unroll
    for (int e2 = 0; e2 < 16; ++e2) Bs[buf][(bn + e2) * 32 + bk] = u.e[e2];
  };

  stage_tile(0, 0);
  wait_async();
  __syncthreads();

  for (int k0 = 0; k0 < K; k0 += 32) {
    int buf = (k0 >> 5) & 1;
    if (k0 + 32 < K) {
      stage_tile(k0 + 32, buf ^ 1);
      if (k0 + 64 < K)  // prefetch the tile after next (global_prefetch_b8)
        __builtin_prefetch(W + (size_t)(k0 + 64 + bk) * N + n0 + bn, 0, 0);
    }
#pragma unroll
    for (int mi = 0; mi < 2; ++mi) {
      v16bf af = load_frag_a(As[buf], 32, wm * 32 + mi * 16, 0, lane);
#pragma unroll
      for (int ni = 0; ni < 4; ++ni) {
        v16bf bfr = load_frag_b(Bs[buf], 32, wn * 64 + ni * 16, 0, lane);
        acc[mi][ni] = __builtin_amdgcn_wmma_f32_16x16x32_bf16(
            false, af, false, bfr, (short)0, acc[mi][ni], false, false);
      }
    }
    wait_async();
    __syncthreads();
  }

  int hi = lane >> 4, cl = lane & 15;
#pragma unroll
  for (int mi = 0; mi < 2; ++mi) {
#pragma unroll
    for (int ni = 0; ni < 4; ++ni) {
      int col = n0 + wn * 64 + ni * 16 + cl;
      float bv = bias[col];
#pragma unroll
      for (int i = 0; i < 8; ++i) {
        int row = m0 + wm * 32 + mi * 16 + i + 8 * hi;
        size_t idx = (size_t)row * N + col;
        float v = acc[mi][ni][i] + bv;
        if (EPI == EPI_F32_BIAS_RES) {
          ((float*)outv)[idx] = v + resid[idx];
        } else if (EPI == EPI_BF16_BIAS_GELU) {
          ((bf16*)outv)[idx] = (bf16)(0.5f * v * (1.f + erff(v * 0.70710678f)));
        } else {
          ((bf16*)outv)[idx] = (bf16)v;
        }
      }
    }
  }
}

// ---------------------------------------------------------------------------
// Sliding-window chunked attention. One block per (chunk, head, batch).
// Q / K window staged via async global->LDS; V transposed manually.
// Waves 0-3: 16 query rows each over the full 128 window + in-reg softmax.
// All 8 waves: P @ V (64x64 out, K=128).
// ---------------------------------------------------------------------------
__global__ __launch_bounds__(256) void attn_kernel(
    const bf16* __restrict__ qb, const bf16* __restrict__ kb,
    const bf16* __restrict__ vb, bf16* __restrict__ ao) {
  __shared__ bf16 Qs[CH * HD];     //  8KB [row][d]
  __shared__ bf16 Ks[WIN * HD];    // 16KB [slot][d]
  __shared__ bf16 Vt[HD * WIN];    // 16KB [d][slot] (transposed)
  __shared__ bf16 Ps[CH * WIN];    // 16KB [row][slot]

  int chunk = blockIdx.x, head = blockIdx.y, batch = blockIdx.z;
  int t = threadIdx.x, lane = t & 31, w = t >> 5;
  size_t rowbase = (size_t)batch * SEQ + (size_t)chunk * CH;
  int hoff = head * HD;

  { // Q: 4 threads per row, 16 bf16 each (async)
    int r = t >> 2, seg = (t & 3) * 16;
    const bf16* src = qb + (rowbase + r) * EMB + hoff + seg;
    bf16* dst = Qs + r * HD + seg;
    async_copy16(src, dst);
    async_copy16(src + 8, dst + 8);
  }
  { // K window: 2 threads per slot, 32 bf16 each; prev chunk of chunk 0 = 0
    int s = t >> 1, seg = (t & 1) * 32;
    int grow = chunk * CH - CH + s;
    bf16* dst = Ks + s * HD + seg;
    if (grow < 0) {
      uint4 z = {0, 0, 0, 0};
      uint4* d4 = (uint4*)dst;
      d4[0] = z; d4[1] = z; d4[2] = z; d4[3] = z;
    } else {
      const bf16* src = kb + ((size_t)batch * SEQ + grow) * EMB + hoff + seg;
#pragma unroll
      for (int j = 0; j < 4; ++j) async_copy16(src + 8 * j, dst + 8 * j);
    }
  }
  { // V window, transposed into [d][slot]
    int s = t >> 1, seg = (t & 1) * 32;
    int grow = chunk * CH - CH + s;
    if (grow < 0) {
#pragma unroll
      for (int e = 0; e < 32; ++e) Vt[(seg + e) * WIN + s] = (bf16)0.f;
    } else {
      union { uint4 q[4]; bf16 e[32]; } u;
      const uint4* src =
          (const uint4*)(vb + ((size_t)batch * SEQ + grow) * EMB + hoff + seg);
      u.q[0] = src[0]; u.q[1] = src[1]; u.q[2] = src[2]; u.q[3] = src[3];
#pragma unroll
      for (int e = 0; e < 32; ++e) Vt[(seg + e) * WIN + s] = u.e[e];
    }
  }
  wait_async();
  __syncthreads();

  if (w < 4) {  // scores + softmax for rows w*16 .. w*16+15
    int mrow = w * 16;
    v8f sc[8];
#pragma unroll
    for (int nt = 0; nt < 8; ++nt) sc[nt] = v8f{};
    for (int k0 = 0; k0 < HD; k0 += 32) {
      v16bf af = load_frag_a(Qs, HD, mrow, k0, lane);
#pragma unroll
      for (int nt = 0; nt < 8; ++nt) {
        v16bf bfr = load_frag_b(Ks, HD, nt * 16, k0, lane);
        sc[nt] = __builtin_amdgcn_wmma_f32_16x16x32_bf16(
            false, af, false, bfr, (short)0, sc[nt], false, false);
      }
    }
    int hi = lane >> 4, cl = lane & 15;
    float mx[8];
#pragma unroll
    for (int i = 0; i < 8; ++i) mx[i] = -3.0e38f;
#pragma unroll
    for (int nt = 0; nt < 8; ++nt) {
      int col = nt * 16 + cl;
#pragma unroll
      for (int i = 0; i < 8; ++i) {
        int rloc = mrow + i + 8 * hi;              // 0..63 within chunk
        float vv = sc[nt][i] * 0.125f;             // 1/sqrt(64)
        if (col > CH + rloc) vv = -3.0e38f;        // causal window mask
        sc[nt][i] = vv;
        mx[i] = fmaxf(mx[i], vv);
      }
    }
#pragma unroll
    for (int i = 0; i < 8; ++i)
      for (int m = 1; m < 16; m <<= 1)
        mx[i] = fmaxf(mx[i], __shfl_xor(mx[i], m, 32));
    float sm[8];
#pragma unroll
    for (int i = 0; i < 8; ++i) sm[i] = 0.f;
#pragma unroll
    for (int nt = 0; nt < 8; ++nt)
#pragma unroll
      for (int i = 0; i < 8; ++i) {
        float e = __expf(sc[nt][i] - mx[i]);
        sc[nt][i] = e;
        sm[i] += e;
      }
#pragma unroll
    for (int i = 0; i < 8; ++i)
      for (int m = 1; m < 16; m <<= 1) sm[i] += __shfl_xor(sm[i], m, 32);
#pragma unroll
    for (int i = 0; i < 8; ++i) sm[i] = 1.f / sm[i];
#pragma unroll
    for (int nt = 0; nt < 8; ++nt)
#pragma unroll
      for (int i = 0; i < 8; ++i)
        Ps[(mrow + i + 8 * hi) * WIN + nt * 16 + cl] =
            (bf16)(sc[nt][i] * sm[i]);
  }
  __syncthreads();

  { // out = P @ V : 16 tiles, 2 per wave
    int mt = (w >> 1) * 16;
    int nbase = (w & 1) * 32;
    v8f o[2] = {v8f{}, v8f{}};
    for (int k0 = 0; k0 < WIN; k0 += 32) {
      v16bf af = load_frag_a(Ps, WIN, mt, k0, lane);
#pragma unroll
      for (int j = 0; j < 2; ++j) {
        v16bf bfr = load_frag_b(Vt, WIN, nbase + j * 16, k0, lane);
        o[j] = __builtin_amdgcn_wmma_f32_16x16x32_bf16(
            false, af, false, bfr, (short)0, o[j], false, false);
      }
    }
    int hi = lane >> 4, cl = lane & 15;
#pragma unroll
    for (int j = 0; j < 2; ++j) {
      int col = hoff + nbase + j * 16 + cl;
#pragma unroll
      for (int i = 0; i < 8; ++i) {
        int row = mt + i + 8 * hi;
        ao[(rowbase + row) * EMB + col] = (bf16)o[j][i];
      }
    }
  }
}

// ---------------------------------------------------------------------------
// Final: duplicate x2 into second output slot, accumulate MSE loss
// ---------------------------------------------------------------------------
__global__ void zero_loss_kernel(float* out, size_t n) {
  if (threadIdx.x == 0 && blockIdx.x == 0) out[2 * n] = 0.f;
}

__global__ __launch_bounds__(256) void loss_copy_kernel(
    float* __restrict__ out, const float* __restrict__ p, size_t n) {
  __shared__ float red[256];
  float acc = 0.f;
  size_t stride = (size_t)gridDim.x * blockDim.x;
  for (size_t i = (size_t)blockIdx.x * blockDim.x + threadIdx.x; i < n;
       i += stride) {
    float v = out[i];
    out[n + i] = v;
    float d = v - p[i];
    acc += d * d;
  }
  red[threadIdx.x] = acc; __syncthreads();
  for (int o = 128; o > 0; o >>= 1) {
    if (threadIdx.x < o) red[threadIdx.x] += red[threadIdx.x + o];
    __syncthreads();
  }
  if (threadIdx.x == 0) atomicAdd(out + 2 * n, red[0] / (float)n);
}

// ---------------------------------------------------------------------------
extern "C" void kernel_launch(void* const* d_in, const int* in_sizes, int n_in,
                              void* d_out, int out_size, void* d_ws,
                              size_t ws_size, hipStream_t stream) {
  (void)in_sizes; (void)n_in; (void)out_size; (void)ws_size;
  const float* x    = (const float*)d_in[0];
  const float* p1   = (const float*)d_in[1];
  const float* wq   = (const float*)d_in[2];
  const float* bq   = (const float*)d_in[3];
  const float* wk   = (const float*)d_in[4];
  const float* bk_  = (const float*)d_in[5];
  const float* wv   = (const float*)d_in[6];
  const float* bv   = (const float*)d_in[7];
  const float* wo   = (const float*)d_in[8];
  const float* bo   = (const float*)d_in[9];
  const float* w1   = (const float*)d_in[10];
  const float* b1   = (const float*)d_in[11];
  const float* w2   = (const float*)d_in[12];
  const float* b2   = (const float*)d_in[13];
  const float* g1   = (const float*)d_in[14];
  const float* be1  = (const float*)d_in[15];
  const float* g2   = (const float*)d_in[16];
  const float* be2  = (const float*)d_in[17];

  const size_t M = TOKENS, E = EMB, F = FFD;
  char* ws = (char*)d_ws;
  size_t off = 0;
  auto alloc = [&](size_t bytes) {
    void* pp = ws + off;
    off += (bytes + 255) & ~(size_t)255;
    return pp;
  };
  bf16* xn   = (bf16*)alloc(M * E * 2);
  bf16* qbuf = (bf16*)alloc(M * E * 2);
  bf16* kbuf = (bf16*)alloc(M * E * 2);
  bf16* vbuf = (bf16*)alloc(M * E * 2);
  bf16* aob  = (bf16*)alloc(M * E * 2);
  bf16* hb   = (bf16*)alloc(M * E * 2);
  float* x1  = (float*)alloc(M * E * 4);
  bf16* ff1  = (bf16*)alloc(M * F * 2);
  bf16* wqb  = (bf16*)alloc(E * E * 2);
  bf16* wkb  = (bf16*)alloc(E * E * 2);
  bf16* wvb  = (bf16*)alloc(E * E * 2);
  bf16* wob  = (bf16*)alloc(E * E * 2);
  bf16* w1b  = (bf16*)alloc(E * F * 2);
  bf16* w2b  = (bf16*)alloc(F * E * 2);

  // 1. weight conversion
  convert_bf16_kernel<<<1024, 256, 0, stream>>>(wq, wqb, E * E);
  convert_bf16_kernel<<<1024, 256, 0, stream>>>(wk, wkb, E * E);
  convert_bf16_kernel<<<1024, 256, 0, stream>>>(wv, wvb, E * E);
  convert_bf16_kernel<<<1024, 256, 0, stream>>>(wo, wob, E * E);
  convert_bf16_kernel<<<2048, 256, 0, stream>>>(w1, w1b, E * F);
  convert_bf16_kernel<<<2048, 256, 0, stream>>>(w2, w2b, F * E);

  // 2. LN1
  ln_kernel<<<M, 256, 0, stream>>>(x, g1, be1, xn);

  // 3. Q/K/V projections
  dim3 gE(E / 128, M / 128);
  gemm_bf16_kernel<EPI_BF16_BIAS><<<gE, 256, 0, stream>>>(
      xn, wqb, bq, nullptr, qbuf, M, E, E);
  gemm_bf16_kernel<EPI_BF16_BIAS><<<gE, 256, 0, stream>>>(
      xn, wkb, bk_, nullptr, kbuf, M, E, E);
  gemm_bf16_kernel<EPI_BF16_BIAS><<<gE, 256, 0, stream>>>(
      xn, wvb, bv, nullptr, vbuf, M, E, E);

  // 4. chunked sliding-window attention
  dim3 gA(SEQ / CH, HEADS, 4);
  attn_kernel<<<gA, 256, 0, stream>>>(qbuf, kbuf, vbuf, aob);

  // 5. output projection + residual -> x1 (f32)
  gemm_bf16_kernel<EPI_F32_BIAS_RES><<<gE, 256, 0, stream>>>(
      aob, wob, bo, x, x1, M, E, E);

  // 6. LN2
  ln_kernel<<<M, 256, 0, stream>>>(x1, g2, be2, hb);

  // 7. FFN1 + GELU
  dim3 gF(F / 128, M / 128);
  gemm_bf16_kernel<EPI_BF16_BIAS_GELU><<<gF, 256, 0, stream>>>(
      hb, w1b, b1, nullptr, ff1, M, F, E);

  // 8. FFN2 + residual -> x2 directly into d_out
  float* outf = (float*)d_out;
  gemm_bf16_kernel<EPI_F32_BIAS_RES><<<gE, 256, 0, stream>>>(
      ff1, w2b, b2, x1, outf, M, E, F);

  // 9. duplicate x2 + MSE loss
  size_t n = M * E;
  zero_loss_kernel<<<1, 64, 0, stream>>>(outf, n);
  loss_copy_kernel<<<4096, 256, 0, stream>>>(outf, p1, n);
}
